// GIN_37048387895934
// MI455X (gfx1250) — compile-verified, matrix operand order checked by reading, not run on hardware
//
#include <hip/hip_runtime.h>

typedef __attribute__((ext_vector_type(16))) _Float16 v16h;
typedef __attribute__((ext_vector_type(8)))  float    v8f;

#define BN_EPS 1e-5f

// K index held by half-slot i (0..15) for lane-half g (0/1), per CDNA5 ISA
// 16-bit A-matrix 16x32 layout (7.12.2).
__device__ __forceinline__ int kmap(int i, int g) {
    return (i < 8) ? (g * 8 + i) : (16 + g * 8 + (i - 8));
}

// ---------------------------------------------------------------------------
// zero fill
__global__ void zero_kernel(float* __restrict__ p, long n) {
    for (long i = blockIdx.x * (long)blockDim.x + threadIdx.x; i < n;
         i += (long)gridDim.x * blockDim.x)
        p[i] = 0.f;
}

// ---------------------------------------------------------------------------
// Pre-pack a (nf x 32) f32 weight matrix into per-lane f16 WMMA B-fragments.
// Output layout: frag[t][lane][i], t = N-tile (0/1), i = half-slot 0..15.
// K rows >= nf are zero (layer-1 K=16 padded to 32).
__global__ void pack_w_kernel(const float* __restrict__ W, int nf,
                              _Float16* __restrict__ out) {
    int t = threadIdx.x >> 5;  // 0..1
    int l = threadIdx.x & 31;
    int gh = l >> 4, n = l & 15;
#pragma unroll
    for (int i = 0; i < 16; ++i) {
        int k = kmap(i, gh);
        float v = (k < nf) ? W[k * 32 + t * 16 + n] : 0.f;
        out[(t * 32 + l) * 16 + i] = (_Float16)v;
    }
}

// ---------------------------------------------------------------------------
// edge scatter: agg[dst] += bn(feat[src]); one thread per (edge, 4-feature group)
__global__ void scatter_kernel(const float* __restrict__ feat,
                               const int* __restrict__ src,
                               const int* __restrict__ dst,
                               float* __restrict__ agg,
                               const float* __restrict__ scale,
                               const float* __restrict__ shift,
                               int use_bn, int nf, int lg, int n_edges) {
    const long total = (long)n_edges << lg;
    const long mask = (1L << lg) - 1;
    for (long t = blockIdx.x * (long)blockDim.x + threadIdx.x; t < total;
         t += (long)gridDim.x * blockDim.x) {
        long e = t >> lg;
        int fg = (int)(t & mask);
        int s = src[e];
        int d = dst[e];
        int c0 = fg << 2;
        float4 v = *(const float4*)(feat + (long)s * nf + c0);
        float vv[4] = {v.x, v.y, v.z, v.w};
        float* ap = agg + (long)d * nf + c0;
#pragma unroll
        for (int k = 0; k < 4; ++k) {
            float val = vv[k];
            if (use_bn) val = fmaf(val, scale[c0 + k], shift[c0 + k]);
            atomicAdd(ap + k, val);
        }
    }
}

// ---------------------------------------------------------------------------
// Fused GIN MLP per 16-node tile (one wave each, 8 waves / block):
//   t = bn(in) + agg ; u = relu(t@Wa + ba) ; h = relu(u@Wb + bb)
// Weights arrive pre-packed as per-lane f16 B fragments.
// Also accumulates per-channel sum / sumsq into stats[0..31]/[32..63].
__global__ void __launch_bounds__(256)
mlp_kernel(const float* __restrict__ in, const float* __restrict__ agg,
           const float* __restrict__ scaleIn, const float* __restrict__ shiftIn,
           int use_bn, int nf_in,
           const _Float16* __restrict__ pWa, const float* __restrict__ ba,
           const _Float16* __restrict__ pWb, const float* __restrict__ bb,
           float* __restrict__ out, float* __restrict__ stats) {
    __shared__ float lds[8][16 * 33];
    const int lane = threadIdx.x & 31;
    const int wv = threadIdx.x >> 5;
    float* T = lds[wv];
    const long node0 = ((long)blockIdx.x * 8 + wv) * 16;

    // ---- stage t = bn(in) + agg into LDS (16 rows x 32 cols, K-padded) ----
    if (nf_in == 32) {
        const int col = lane;
        const float sc = use_bn ? scaleIn[col] : 1.f;
        const float sh = use_bn ? shiftIn[col] : 0.f;
#pragma unroll 4
        for (int r = 0; r < 16; ++r) {
            long idx = (node0 + r) * 32 + col;
            T[r * 33 + col] = fmaf(in[idx], sc, sh) + agg[idx];
        }
    } else {  // nf_in == 16, no BN (layer 1); zero-pad K 16..31
        const int col = lane & 15;
        const int rh = lane >> 4;
#pragma unroll 4
        for (int r2 = 0; r2 < 8; ++r2) {
            int r = r2 * 2 + rh;
            long idx = (node0 + r) * 16 + col;
            T[r * 33 + col] = in[idx] + agg[idx];
            T[r * 33 + 16 + col] = 0.f;
        }
    }
    __syncthreads();

    const int m = lane & 15;   // row (A) / col (B,C,D)
    const int gh = lane >> 4;  // lane-half
    const v16h* pa = (const v16h*)pWa;
    const v16h* pb = (const v16h*)pWb;

    // ---- GEMM 1: u = relu(t @ Wa + ba), two 16-col tiles ----
    v16h a;
#pragma unroll
    for (int i = 0; i < 16; ++i) a[i] = (_Float16)T[m * 33 + kmap(i, gh)];

    v8f c0, c1;
    {
        v16h b0 = pa[lane];
        v16h b1 = pa[32 + lane];
        const float bias0 = ba[m], bias1 = ba[16 + m];
#pragma unroll
        for (int j = 0; j < 8; ++j) { c0[j] = bias0; c1[j] = bias1; }
        c0 = __builtin_amdgcn_wmma_f32_16x16x32_f16(false, a, false, b0,
                                                    (short)0, c0, false, false);
        c1 = __builtin_amdgcn_wmma_f32_16x16x32_f16(false, a, false, b1,
                                                    (short)0, c1, false, false);
    }
    __syncthreads();

    // ---- relu(u) back to LDS (transpose to A layout) ----
#pragma unroll
    for (int j = 0; j < 8; ++j) {
        T[(j + 8 * gh) * 33 + m] = fmaxf(c0[j], 0.f);
        T[(j + 8 * gh) * 33 + 16 + m] = fmaxf(c1[j], 0.f);
    }
    __syncthreads();

    // ---- GEMM 2: h = relu(u @ Wb + bb) ----
    v16h a2;
#pragma unroll
    for (int i = 0; i < 16; ++i) a2[i] = (_Float16)T[m * 33 + kmap(i, gh)];

    v8f d0, d1;
    {
        v16h b0 = pb[lane];
        v16h b1 = pb[32 + lane];
        const float bias0 = bb[m], bias1 = bb[16 + m];
#pragma unroll
        for (int j = 0; j < 8; ++j) { d0[j] = bias0; d1[j] = bias1; }
        d0 = __builtin_amdgcn_wmma_f32_16x16x32_f16(false, a2, false, b0,
                                                    (short)0, d0, false, false);
        d1 = __builtin_amdgcn_wmma_f32_16x16x32_f16(false, a2, false, b1,
                                                    (short)0, d1, false, false);
    }

    // ---- relu + store h + BN statistics ----
    float s0 = 0.f, q0 = 0.f, s1 = 0.f, q1 = 0.f;
#pragma unroll
    for (int j = 0; j < 8; ++j) {
        float h0 = fmaxf(d0[j], 0.f);
        float h1 = fmaxf(d1[j], 0.f);
        long row = node0 + j + 8 * gh;
        out[row * 32 + m] = h0;
        out[row * 32 + 16 + m] = h1;
        s0 += h0; q0 += h0 * h0;
        s1 += h1; q1 += h1 * h1;
    }
    s0 += __shfl_xor(s0, 16, 32);  q0 += __shfl_xor(q0, 16, 32);
    s1 += __shfl_xor(s1, 16, 32);  q1 += __shfl_xor(q1, 16, 32);
    if (lane < 16) {
        atomicAdd(&stats[lane], s0);
        atomicAdd(&stats[16 + lane], s1);
        atomicAdd(&stats[32 + lane], q0);
        atomicAdd(&stats[48 + lane], q1);
    }
}

// ---------------------------------------------------------------------------
// stats -> per-channel BN scale/shift (population variance, ddof=0)
__global__ void bn_finalize(const float* __restrict__ st,
                            const float* __restrict__ g,
                            const float* __restrict__ be,
                            float* __restrict__ sc, float* __restrict__ sh,
                            float inv_n) {
    int c = threadIdx.x;
    float mean = st[c] * inv_n;
    float var = st[32 + c] * inv_n - mean * mean;
    float r = rsqrtf(var + BN_EPS);
    float s = g[c] * r;
    sc[c] = s;
    sh[c] = be[c] - mean * s;
}

// ---------------------------------------------------------------------------
// per-graph: sum-pool 64 nodes (BN folded: linear), 32->2 linear, log_softmax
__global__ void final_kernel(const float* __restrict__ h,
                             const float* __restrict__ scale,
                             const float* __restrict__ shift,
                             const float* __restrict__ Wf,
                             const float* __restrict__ bf,
                             float* __restrict__ out, int n_graphs) {
    int lane = threadIdx.x & 31;
    int wv = threadIdx.x >> 5;
    int g = blockIdx.x * 8 + wv;
    if (g >= n_graphs) return;
    float s = 0.f;
    const float* hp = h + (long)g * 64 * 32 + lane;
#pragma unroll 8
    for (int n = 0; n < 64; ++n) s += hp[n * 32];
    float p = fmaf(s, scale[lane], 64.f * shift[lane]);
    float l0 = p * Wf[lane * 2 + 0];
    float l1 = p * Wf[lane * 2 + 1];
#pragma unroll
    for (int off = 16; off > 0; off >>= 1) {
        l0 += __shfl_xor(l0, off, 32);
        l1 += __shfl_xor(l1, off, 32);
    }
    if (lane == 0) {
        l0 += bf[0];
        l1 += bf[1];
        float mx = fmaxf(l0, l1);
        float lse = mx + logf(expf(l0 - mx) + expf(l1 - mx));
        out[(long)g * 2 + 0] = l0 - lse;
        out[(long)g * 2 + 1] = l1 - lse;
    }
}

// ---------------------------------------------------------------------------
extern "C" void kernel_launch(void* const* d_in, const int* in_sizes, int n_in,
                              void* d_out, int out_size, void* d_ws, size_t ws_size,
                              hipStream_t stream) {
    const float* x = (const float*)d_in[0];
    const int* ei = (const int*)d_in[1];
    const float* W1a = (const float*)d_in[3];
    const float* b1a = (const float*)d_in[4];
    const float* W1b = (const float*)d_in[5];
    const float* b1b = (const float*)d_in[6];
    const float* W2a = (const float*)d_in[7];
    const float* b2a = (const float*)d_in[8];
    const float* W2b = (const float*)d_in[9];
    const float* b2b = (const float*)d_in[10];
    const float* W3a = (const float*)d_in[11];
    const float* b3a = (const float*)d_in[12];
    const float* W3b = (const float*)d_in[13];
    const float* b3b = (const float*)d_in[14];
    const float* g1 = (const float*)d_in[15];
    const float* be1 = (const float*)d_in[16];
    const float* g2 = (const float*)d_in[17];
    const float* be2 = (const float*)d_in[18];
    const float* g3 = (const float*)d_in[19];
    const float* be3 = (const float*)d_in[20];
    const float* Wf = (const float*)d_in[21];
    const float* bf = (const float*)d_in[22];

    const int n_nodes = in_sizes[0] / 16;
    const int n_edges = in_sizes[1] / 2;
    const int n_graphs = n_nodes / 64;
    const int* src = ei;
    const int* dst = ei + n_edges;

    float* ws = (float*)d_ws;
    const size_t per = (size_t)n_nodes * 32;
    float* buf0 = ws;
    float* buf1 = ws + per;
    float* buf2 = ws + 2 * per;
    float* aux = ws + 3 * per;
    float* st1 = aux;        float* sc1 = aux + 64;  float* sh1 = aux + 96;
    float* st2 = aux + 128;  float* sc2 = aux + 192; float* sh2 = aux + 224;
    float* st3 = aux + 256;  float* sc3 = aux + 320; float* sh3 = aux + 352;

    // pre-packed f16 B fragments: 1024 halves (2 KB) per weight matrix
    _Float16* pk = (_Float16*)(aux + 512);
    _Float16* pW1a = pk;
    _Float16* pW1b = pk + 1024;
    _Float16* pW2a = pk + 2048;
    _Float16* pW2b = pk + 3072;
    _Float16* pW3a = pk + 4096;
    _Float16* pW3b = pk + 5120;

    pack_w_kernel<<<1, 64, 0, stream>>>(W1a, 16, pW1a);
    pack_w_kernel<<<1, 64, 0, stream>>>(W1b, 32, pW1b);
    pack_w_kernel<<<1, 64, 0, stream>>>(W2a, 32, pW2a);
    pack_w_kernel<<<1, 64, 0, stream>>>(W2b, 32, pW2b);
    pack_w_kernel<<<1, 64, 0, stream>>>(W3a, 32, pW3a);
    pack_w_kernel<<<1, 64, 0, stream>>>(W3b, 32, pW3b);

    const int zblocks = 4096;
    const int mlpBlocks = n_nodes / 128;  // 8 waves x 16 nodes per block
    const float inv_n = 1.f / (float)n_nodes;

    auto layer = [&](const float* in, float* agg, float* out, int nf,
                     const float* scIn, const float* shIn, int bn,
                     const _Float16* pWa, const float* ba_,
                     const _Float16* pWb, const float* bb_, float* st,
                     const float* gam, const float* bet, float* sc, float* sh) {
        zero_kernel<<<zblocks, 256, 0, stream>>>(agg, (long)n_nodes * nf);
        zero_kernel<<<1, 64, 0, stream>>>(st, 64);
        int lg = (nf == 32) ? 3 : 2;
        long total = (long)n_edges << lg;
        int sblocks = (int)((total + 255) / 256);
        scatter_kernel<<<sblocks, 256, 0, stream>>>(in, src, dst, agg, scIn,
                                                    shIn, bn, nf, lg, n_edges);
        mlp_kernel<<<mlpBlocks, 256, 0, stream>>>(in, agg, scIn, shIn, bn, nf,
                                                  pWa, ba_, pWb, bb_, out, st);
        bn_finalize<<<1, 32, 0, stream>>>(st, gam, bet, sc, sh, inv_n);
    };

    // layer 1: in = x (16 feats, no BN), agg in buf0, h1 -> buf1
    layer(x, buf0, buf1, 16, nullptr, nullptr, 0, pW1a, b1a, pW1b, b1b,
          st1, g1, be1, sc1, sh1);
    // layer 2: in = h1 (BN1 folded), agg in buf2, h2 -> buf0
    layer(buf1, buf2, buf0, 32, sc1, sh1, 1, pW2a, b2a, pW2b, b2b,
          st2, g2, be2, sc2, sh2);
    // layer 3: in = h2 (BN2 folded), agg in buf1, h3 -> buf2
    layer(buf0, buf1, buf2, 32, sc2, sh2, 1, pW3a, b3a, pW3b, b3b,
          st3, g3, be3, sc3, sh3);

    // pool + classifier + log_softmax (BN3 folded linearly into pooling)
    final_kernel<<<(n_graphs + 7) / 8, 256, 0, stream>>>(buf2, sc3, sh3, Wf,
                                                         bf, (float*)d_out,
                                                         n_graphs);
}
